// AttentionAggregator_47261820125624
// MI455X (gfx1250) — matrix operand order repaired and checked by reading, not compiled, not run on hardware
//
#include <hip/hip_runtime.h>
#include <stdint.h>

// ---------------- problem constants ----------------
#define T_DIM   5
#define D_DIM   128
#define BT      16               // batches per block
#define ROWS    (BT * T_DIM)     // 80 x-rows per block

// LDS pitches (elements): 16B-aligned rows, bank-conflict-free fragments
#define X32P    132              // f32 x rows  (b,t)
#define X16P    136              // f16 x rows
#define MP      136              // f16 mean rows
#define WTP     264              // f16 W^T rows (n): 256 + pad

typedef _Float16 h8  __attribute__((ext_vector_type(8)));
typedef _Float16 h16 __attribute__((ext_vector_type(16)));
typedef float    f4  __attribute__((ext_vector_type(4)));
typedef float    f8v __attribute__((ext_vector_type(8)));
typedef float    v8f __attribute__((ext_vector_type(8)));
typedef int      v4i __attribute__((vector_size(16)));   // matches builtin param

#if defined(__has_builtin)
#if __has_builtin(__builtin_amdgcn_global_load_async_to_lds_b128)
#define HAVE_ASYNC_LDS 1
#endif
#endif

__device__ __forceinline__ void async_cp16(const void* g, void* l) {
#ifdef HAVE_ASYNC_LDS
  __builtin_amdgcn_global_load_async_to_lds_b128((v4i*)g, (v4i*)l, 0, 0);
#else
  *(f4*)l = *(const f4*)g;
#endif
}

__device__ __forceinline__ void wait_async() {
#ifdef HAVE_ASYNC_LDS
#if __has_builtin(__builtin_amdgcn_s_wait_asynccnt)
  __builtin_amdgcn_s_wait_asynccnt(0);
#else
  asm volatile("s_wait_asynccnt 0" ::: "memory");
#endif
#endif
}

__device__ __forceinline__ h16 cat8(h8 a, h8 b) {
  return __builtin_shufflevector(a, b, 0,1,2,3,4,5,6,7,8,9,10,11,12,13,14,15);
}

__device__ __forceinline__ f8v load8(const float* p) {
  f4 a = *(const f4*)p, b = *(const f4*)(p + 4);
  return __builtin_shufflevector(a, b, 0,1,2,3,4,5,6,7);
}

// ---- one-time prep: wt[n*256+k] = (_Float16)(w[k*128+n] + bias), 64 KB ----
__global__ __launch_bounds__(256)
void prep_wT(const float* __restrict__ w, const float* __restrict__ biasp,
             _Float16* __restrict__ wt) {
  const int g = blockIdx.x * 256 + threadIdx.x;   // 0..32767
  const int n = g & 127;
  const int k = g >> 7;                           // 0..255 (coalesced reads)
  wt[n * 256 + k] = (_Float16)(w[(long)k * D_DIM + n] + biasp[0]);
}

__global__ __launch_bounds__(256) __attribute__((amdgpu_waves_per_eu(4)))
void attn_agg_wmma(const float* __restrict__ x, const float* __restrict__ w,
                   const float* __restrict__ biasp, float* __restrict__ out,
                   const _Float16* __restrict__ wt) {
  __shared__ alignas(16) float    x32[ROWS * X32P];   // 42240 B
  __shared__ alignas(16) _Float16 x16[ROWS * X16P];   // 21760 B
  __shared__ alignas(16) _Float16 m16[BT * MP];       //  4352 B
  __shared__ alignas(16) _Float16 wT [D_DIM * WTP];   // 67584 B
  __shared__ float normsq[BT];

  const int tid  = threadIdx.x;
  const int wave = tid >> 5;
  const int lane = tid & 31;
  const int l16  = lane & 15;
  const int lh   = lane >> 4;           // which 16-lane half
  const long b0  = (long)blockIdx.x * BT;

  if (tid < BT) normsq[tid] = 0.0f;

  // ---- stage x tile (80 rows * 512B, contiguous) into LDS, async ----
  {
    const float* gbase = x + (b0 * T_DIM) * D_DIM;
#pragma unroll
    for (int i = 0; i < 10; ++i) {
      int row = wave + i * 8;
      async_cp16(gbase + (long)row * D_DIM + lane * 4,
                 x32 + row * X32P + lane * 4);
    }
  }

  // ---- stage prebuilt f16 W^T (64KB) into padded LDS rows, async ----
  if (wt) {
#pragma unroll
    for (int i = 0; i < 16; ++i) {
      int c = tid + i * 256;            // 4096 16B-chunks, 32 per row
      int row = c >> 5;
      int col = (c & 31) << 3;          // halves
      async_cp16(wt + row * 256 + col, wT + row * WTP + col);
    }
  } else {
    const float bias = biasp[0];
    const int n  = tid & 127;
    const int kh = tid >> 7;
#pragma unroll 8
    for (int j = 0; j < 128; ++j) {
      int k = kh * 128 + j;
      wT[n * WTP + k] = (_Float16)(w[(long)k * D_DIM + n] + bias);
    }
  }

  wait_async();
  __syncthreads();

  // ---- f16 copy of x (8-wide vector cvt), and per-batch means ----
  for (int c = tid; c < ROWS * 16; c += 256) {     // 5 iterations
    int row = c >> 4, k8 = (c & 15) << 3;
    f8v v = load8(x32 + row * X32P + k8);
    *(h8*)(x16 + row * X16P + k8) = __builtin_convertvector(v, h8);
  }
  {                                                 // exactly 1 chunk/thread
    int b = tid >> 4, k8 = (tid & 15) << 3;
    f8v s = load8(x32 + (b * T_DIM) * X32P + k8);
#pragma unroll
    for (int t = 1; t < T_DIM; ++t) s += load8(x32 + (b * T_DIM + t) * X32P + k8);
    s *= 0.2f;
    *(h8*)(m16 + b * MP + k8) = __builtin_convertvector(s, h8);
  }
  __syncthreads();

  // ---- WMMA: wave owns output columns [nb, nb+16) ----
  const int nb = wave * 16;
  const int nIdx = nb + l16;            // this lane's B-matrix column

  v8f acc[T_DIM];
#pragma unroll
  for (int t = 0; t < T_DIM; ++t) acc[t] = (v8f){0,0,0,0,0,0,0,0};
  v8f accM = (v8f){0,0,0,0,0,0,0,0};

  // x @ W1  (K = 0..127), M-tile == t, M-row == local batch
#pragma unroll
  for (int kk = 0; kk < 4; ++kk) {
    const int kb = kk * 32;
    const _Float16* bp = wT + nIdx * WTP + kb + lh * 8;
    h16 bf = cat8(*(const h8*)bp, *(const h8*)(bp + 16));
#pragma unroll
    for (int t = 0; t < T_DIM; ++t) {
      const _Float16* ap = x16 + (l16 * T_DIM + t) * X16P + kb + lh * 8;
      h16 af = cat8(*(const h8*)ap, *(const h8*)(ap + 16));
      acc[t] = __builtin_amdgcn_wmma_f32_16x16x32_f16(
          false, af, false, bf, (short)0, acc[t], false, false);
    }
  }
  // mean @ W2 (K = 128..255), t-invariant -> single accumulator
#pragma unroll
  for (int kk = 0; kk < 4; ++kk) {
    const int kb = kk * 32;
    const _Float16* bp = wT + nIdx * WTP + 128 + kb + lh * 8;
    h16 bf = cat8(*(const h8*)bp, *(const h8*)(bp + 16));
    const _Float16* ap = m16 + l16 * MP + kb + lh * 8;
    h16 af = cat8(*(const h8*)ap, *(const h8*)(ap + 16));
    accM = __builtin_amdgcn_wmma_f32_16x16x32_f16(
        false, af, false, bf, (short)0, accM, false, false);
  }

  // ---- softmax over t (per-lane), weighted sum, sumsq reduction ----
  const int dl = nb + l16;              // output column of this lane
  float y[8];
#pragma unroll
  for (int v = 0; v < 8; ++v) {
    const int bl = v + (lh << 3);       // local batch of this C slot
    float a[T_DIM];
#pragma unroll
    for (int t = 0; t < T_DIM; ++t) a[t] = acc[t][v] + accM[v];
    float mx = a[0];
#pragma unroll
    for (int t = 1; t < T_DIM; ++t) mx = fmaxf(mx, a[t]);
    float e[T_DIM], s = 0.0f;
#pragma unroll
    for (int t = 0; t < T_DIM; ++t) { e[t] = __expf(a[t] - mx); s += e[t]; }
    const float inv = 1.0f / s;
    float yv = 0.0f;
#pragma unroll
    for (int t = 0; t < T_DIM; ++t)
      yv += e[t] * inv * x32[(bl * T_DIM + t) * X32P + dl];
    y[v] = yv;

    float sq = yv * yv;                 // reduce over the 16 lanes sharing bl
    sq += __shfl_xor(sq, 1, 16);
    sq += __shfl_xor(sq, 2, 16);
    sq += __shfl_xor(sq, 4, 16);
    sq += __shfl_xor(sq, 8, 16);
    if (l16 == 0) atomicAdd(&normsq[bl], sq);
  }
  __syncthreads();

  // ---- L2 normalize and store ----
#pragma unroll
  for (int v = 0; v < 8; ++v) {
    const int bl = v + (lh << 3);
    const float denom = fmaxf(sqrtf(normsq[bl]), 1e-12f);
    out[(b0 + bl) * D_DIM + dl] = y[v] / denom;
  }
}

extern "C" void kernel_launch(void* const* d_in, const int* in_sizes, int n_in,
                              void* d_out, int out_size, void* d_ws, size_t ws_size,
                              hipStream_t stream) {
  const float* x     = (const float*)d_in[0];
  const float* w     = (const float*)d_in[1];
  const float* biasp = (const float*)d_in[2];
  float* out = (float*)d_out;

  const int batches = in_sizes[0] / (T_DIM * D_DIM);   // 65536
  const bool useWs = (ws_size >= (size_t)(2 * D_DIM * 256)); // 64 KB f16 W^T

  if (useWs) {
    prep_wT<<<dim3(128), dim3(256), 0, stream>>>(w, biasp, (_Float16*)d_ws);
  }
  attn_agg_wmma<<<dim3(batches / BT), dim3(256), 0, stream>>>(
      x, w, biasp, out, useWs ? (const _Float16*)d_ws : nullptr);
}
